// LorentzMultiHeadedAttention_5205500363345
// MI455X (gfx1250) — matrix operand order repaired and checked by reading, not compiled, not run on hardware
//
#include <hip/hip_runtime.h>
#include <hip/hip_bf16.h>

typedef float  v2f  __attribute__((ext_vector_type(2)));
typedef float  v4f  __attribute__((ext_vector_type(4)));
typedef float  v8f  __attribute__((ext_vector_type(8)));
typedef __bf16 v16bf __attribute__((ext_vector_type(16)));
typedef __bf16 v8bf  __attribute__((ext_vector_type(8)));
typedef __bf16 v4bf  __attribute__((ext_vector_type(4)));

union BF16x16 { v16bf v; v8bf h[2]; };

#define SEQ   1024
#define DIM   512
#define NH    8
#define DPH   64
#define BATCH 16
#define NROWS (BATCH * SEQ)   // 16384

// fp32 exact path: D = A(16x4) * B(4x16) + C
__device__ __forceinline__ v8f wmma4(v2f a, v2f b, v8f c) {
    return __builtin_amdgcn_wmma_f32_16x16x4_f32(
        false, a, false, b, (short)0, c, false, false);
}
// bf16 path: D = A(16x32) * B(32x16) + C, f32 accumulate
__device__ __forceinline__ v8f wmma_bf16(v16bf a, v16bf b, v8f c) {
    return __builtin_amdgcn_wmma_f32_16x16x32_bf16(
        false, a, false, b, (short)0, c, false, false);
}

// ---------------------------------------------------------------------------
// Kernel 0: split fp32 weight into bf16 hi/lo planes (run once per W).
// ---------------------------------------------------------------------------
__global__ void convert_w_kernel(const float* __restrict__ W,
                                 __bf16* __restrict__ hi,
                                 __bf16* __restrict__ lo)
{
    const int i = (blockIdx.x * 256 + threadIdx.x) * 4;   // DIM*DIM elements
    v4f x = *(const v4f*)(W + i);
    v4bf hv, lv;
    #pragma unroll
    for (int j = 0; j < 4; ++j) {
        __bf16 h = (__bf16)x[j];
        hv[j] = h;
        lv[j] = (__bf16)(x[j] - (float)h);
    }
    *(v4bf*)(hi + i) = hv;
    *(v4bf*)(lo + i) = lv;
}

// ---------------------------------------------------------------------------
// Kernel 1: Lorentz linear projection, split-bf16 GEMM (hi*hi + hi*lo + lo*hi).
// Block = 256 threads (8 waves): 32 rows x 512 cols. Wave w owns cols
// [w*64, w*64+64) and both 16-row tiles. X K-halves staged to LDS as bf16
// hi/lo planes; W read from pre-split bf16 planes (L2 resident).
// ---------------------------------------------------------------------------
__launch_bounds__(256, 2)
__global__ void lorentz_proj_kernel(const float* __restrict__ X,
                                    const __bf16* __restrict__ Whi,
                                    const __bf16* __restrict__ Wlo,
                                    const float* __restrict__ bias,
                                    const float* __restrict__ lsptr,
                                    float* __restrict__ Y)
{
    __shared__ __bf16 Xh[32 * 264];   // 256 K + 8 pad (stride 528B, 16B-mult)
    __shared__ __bf16 Xl[32 * 264];
    __shared__ float Psh[8 * 32];
    __shared__ float y0sh[32];
    __shared__ float timesh[32];
    __shared__ float sclsh[32];

    const int t    = threadIdx.x;
    const int wave = t >> 5;
    const int lane = t & 31;
    const int l16  = lane & 15;
    const int half = lane >> 4;
    const int rowbase = blockIdx.x * 32;
    const int colbase = wave * 64;

    v8f acc[2][4] = {};

    for (int kh = 0; kh < DIM; kh += 256) {
        __syncthreads();   // protect LDS reads of previous half
        // Stage + split X[32 x 256] into bf16 hi/lo planes.
        #pragma unroll
        for (int i = 0; i < 8; ++i) {
            int idx = t + i * 256;          // 2048 float4 slots
            int row = idx >> 6;             // 64 float4 per row
            int c4  = (idx & 63) * 4;
            v4f v = *(const v4f*)(X + (size_t)(rowbase + row) * DIM + kh + c4);
            v4bf hv, lv;
            #pragma unroll
            for (int j = 0; j < 4; ++j) {
                __bf16 h = (__bf16)v[j];
                hv[j] = h;
                lv[j] = (__bf16)(v[j] - (float)h);
            }
            *(v4bf*)(Xh + row * 264 + c4) = hv;
            *(v4bf*)(Xl + row * 264 + c4) = lv;
        }
        __syncthreads();

        for (int k0 = 0; k0 < 256; k0 += 32) {
            BF16x16 ah[2], al[2];
            #pragma unroll
            for (int g = 0; g < 2; ++g) {
                const int arow = (g * 16 + l16) * 264;
                const int ka   = k0 + 8 * half;        // K 0..7 / 8..15 group
                ah[g].h[0] = *(const v8bf*)(Xh + arow + ka);
                ah[g].h[1] = *(const v8bf*)(Xh + arow + ka + 16);
                al[g].h[0] = *(const v8bf*)(Xl + arow + ka);
                al[g].h[1] = *(const v8bf*)(Xl + arow + ka + 16);
            }
            #pragma unroll
            for (int f = 0; f < 4; ++f) {
                const size_t wof = (size_t)(colbase + f * 16 + l16) * DIM
                                 + kh + k0 + 8 * half; // K 0..7 / 8..15 group
                BF16x16 bh_, bl_;
                bh_.h[0] = *(const v8bf*)(Whi + wof);
                bh_.h[1] = *(const v8bf*)(Whi + wof + 16);
                bl_.h[0] = *(const v8bf*)(Wlo + wof);
                bl_.h[1] = *(const v8bf*)(Wlo + wof + 16);
                #pragma unroll
                for (int g = 0; g < 2; ++g) {
                    acc[g][f] = wmma_bf16(ah[g].v, bh_.v, acc[g][f]);
                    acc[g][f] = wmma_bf16(ah[g].v, bl_.v, acc[g][f]);
                    acc[g][f] = wmma_bf16(al[g].v, bh_.v, acc[g][f]);
                }
            }
        }
    }

    const float ls = lsptr[0];
    const float escale = expf(ls);

    // Bias + per-row sum of rest^2 (global col 0 excluded, raw value stashed).
    float p[2][8];
    #pragma unroll
    for (int g = 0; g < 2; ++g)
        #pragma unroll
        for (int r = 0; r < 8; ++r) p[g][r] = 0.f;

    #pragma unroll
    for (int f = 0; f < 4; ++f) {
        const int col = colbase + f * 16 + l16;
        const float bv = bias[col];
        #pragma unroll
        for (int g = 0; g < 2; ++g)
            #pragma unroll
            for (int r = 0; r < 8; ++r) {
                float v = acc[g][f][r] + bv;
                acc[g][f][r] = v;
                float sq = v * v;
                if (col == 0) {
                    sq = 0.f;
                    y0sh[g * 16 + r + 8 * half] = v;
                }
                p[g][r] += sq;
            }
    }
    #pragma unroll
    for (int m = 1; m <= 8; m <<= 1)
        #pragma unroll
        for (int g = 0; g < 2; ++g)
            #pragma unroll
            for (int r = 0; r < 8; ++r)
                p[g][r] += __shfl_xor(p[g][r], m, 32);

    if (l16 == 0) {
        #pragma unroll
        for (int g = 0; g < 2; ++g)
            #pragma unroll
            for (int r = 0; r < 8; ++r)
                Psh[wave * 32 + g * 16 + r + 8 * half] = p[g][r];
    }
    __syncthreads();

    if (t < 32) {
        float sq = 0.f;
        #pragma unroll
        for (int w2 = 0; w2 < 8; ++w2) sq += Psh[w2 * 32 + t];
        sq = fmaxf(sq, 1e-8f);
        const float y0   = y0sh[t];
        const float time = escale / (1.f + expf(-y0)) + 1.0f + 1e-4f;
        timesh[t] = time;
        sclsh[t]  = sqrtf((time * time - 1.0f) / sq);
    }
    __syncthreads();

    #pragma unroll
    for (int g = 0; g < 2; ++g)
        #pragma unroll
        for (int f = 0; f < 4; ++f)
            #pragma unroll
            for (int r = 0; r < 8; ++r) {
                const int row = g * 16 + r + 8 * half;
                const int col = colbase + f * 16 + l16;
                const float val = (col == 0) ? timesh[row]
                                             : acc[g][f][r] * sclsh[row];
                Y[(size_t)(rowbase + row) * DIM + col] = val;
            }
}

// ---------------------------------------------------------------------------
// Kernel 2: partial second moment M = K^T V and vsum, S split 4 ways for
// occupancy (512 blocks). Exact fp32 WMMA.
// ---------------------------------------------------------------------------
__launch_bounds__(256, 2)
__global__ void kv_moment_kernel(const float* __restrict__ Kb,
                                 const float* __restrict__ Vb,
                                 float* __restrict__ Mpart,
                                 float* __restrict__ vspart)
{
    __shared__ float vs[4][64];
    const int blk   = blockIdx.x;
    const int chunk = blk >> 7;       // 0..3
    const int bh    = blk & 127;
    const int b  = bh >> 3;
    const int h  = bh & 7;
    const size_t base = (size_t)b * SEQ * DIM + (size_t)h * DPH;
    const int sbeg = chunk * 256;

    const int t = threadIdx.x;
    {   // vsum partial over this 256-row S chunk
        const int d  = t & 63;
        const int sl = t >> 6;
        float a = 0.f;
        for (int s = sbeg + sl * 64; s < sbeg + sl * 64 + 64; ++s)
            a += Vb[base + (size_t)s * DIM + d];
        vs[sl][d] = a;
    }

    const int wave = t >> 5;
    const int lane = t & 31;
    const int l16  = lane & 15;
    const int half = lane >> 4;
    const int dr   = wave & 3;
    const int ec0  = (wave >> 2) * 2;

    v8f acc[2] = {};
    for (int s0 = sbeg; s0 < sbeg + 256; s0 += 4) {
        const int sa = s0 + 2 * half;
        v2f a;                                           // A[m=d, k=s] = K[s,d]
        a.x = Kb[base + (size_t)(sa + 0) * DIM + dr * 16 + l16];
        a.y = Kb[base + (size_t)(sa + 1) * DIM + dr * 16 + l16];
        #pragma unroll
        for (int e = 0; e < 2; ++e) {
            v2f bb;                                      // B[k=s, n=e] = V[s,e]
            bb.x = Vb[base + (size_t)(sa + 0) * DIM + (ec0 + e) * 16 + l16];
            bb.y = Vb[base + (size_t)(sa + 1) * DIM + (ec0 + e) * 16 + l16];
            acc[e] = wmma4(a, bb, acc[e]);
        }
    }
    float* Mout = Mpart + (size_t)(chunk * 128 + bh) * 4096;
    #pragma unroll
    for (int e = 0; e < 2; ++e)
        #pragma unroll
        for (int r = 0; r < 8; ++r)
            Mout[(dr * 16 + r + 8 * half) * 64 + (ec0 + e) * 16 + l16] =
                acc[e][r];

    __syncthreads();
    if (t < 64)
        vspart[(size_t)(chunk * 128 + bh) * 64 + t] =
            vs[0][t] + vs[1][t] + vs[2][t] + vs[3][t];
}

// ---------------------------------------------------------------------------
// Kernel 3: ave = c1*(q o sgn) @ M + c2*vsum ; ctx = ave / sqrt(|sum sgn*ave^2|)
// Sums the 4 M/vsum partials while staging into LDS. Exact fp32 WMMA.
// ---------------------------------------------------------------------------
__launch_bounds__(256, 2)
__global__ void ctx_kernel(const float* __restrict__ Qb,
                           const float* __restrict__ Mpart,
                           const float* __restrict__ vspart,
                           const float* __restrict__ asptr,
                           const float* __restrict__ abptr,
                           float* __restrict__ Out)
{
    __shared__ float Mlds[64 * 68];
    __shared__ float vsl[64];

    const int blk    = blockIdx.x;
    const int bh     = blk >> 3;
    const int schunk = blk & 7;
    const int b = bh >> 3;
    const int h = bh & 7;

    const int t = threadIdx.x;
    #pragma unroll
    for (int i = 0; i < 4; ++i) {
        int idx = t + i * 256;          // 1024 float4 slots
        int row = idx >> 4;
        int c4  = (idx & 15) * 4;
        v4f s = {};
        #pragma unroll
        for (int ch = 0; ch < 4; ++ch) {
            v4f m = *(const v4f*)(Mpart + (size_t)(ch * 128 + bh) * 4096
                                  + row * 64 + c4);
            s += m;
        }
        *(v4f*)(Mlds + row * 68 + c4) = s;
    }
    if (t < 64) {
        float s = 0.f;
        #pragma unroll
        for (int ch = 0; ch < 4; ++ch)
            s += vspart[(size_t)(ch * 128 + bh) * 64 + t];
        vsl[t] = s;
    }
    __syncthreads();

    const float as = asptr[0];
    const float ab = abptr[0];
    const float c1 = 2.f / as;
    const float c2 = c1 + ab;

    const int wave = t >> 5;
    const int lane = t & 31;
    const int l16  = lane & 15;
    const int half = lane >> 4;

    const int s0 = schunk * 128 + wave * 16;
    const size_t qrow = (size_t)b * SEQ * DIM + (size_t)(s0 + l16) * DIM
                        + (size_t)h * DPH;

    v8f acc[4] = {};
    for (int k0 = 0; k0 < DPH; k0 += 4) {
        const int kk = k0 + 2 * half;
        v2f a = *(const v2f*)(Qb + qrow + kk);  // A[m=s-row, k=d]
        if (kk == 0) a.x = -a.x;                // sgn on d==0
        #pragma unroll
        for (int f = 0; f < 4; ++f) {
            v2f bb;                             // B[k=d, n=e] = M[d][e]
            bb.x = Mlds[(kk + 0) * 68 + f * 16 + l16];
            bb.y = Mlds[(kk + 1) * 68 + f * 16 + l16];
            acc[f] = wmma4(a, bb, acc[f]);
        }
    }

    float p[8];
    #pragma unroll
    for (int r = 0; r < 8; ++r) p[r] = 0.f;

    #pragma unroll
    for (int f = 0; f < 4; ++f) {
        const float vsv = vsl[f * 16 + l16];
        #pragma unroll
        for (int r = 0; r < 8; ++r) {
            float av = c1 * acc[f][r] + c2 * vsv;
            acc[f][r] = av;
            float sq = av * av;
            if (f == 0 && l16 == 0) sq = -sq;   // sgn on d==0
            p[r] += sq;
        }
    }
    #pragma unroll
    for (int m = 1; m <= 8; m <<= 1)
        #pragma unroll
        for (int r = 0; r < 8; ++r)
            p[r] += __shfl_xor(p[r], m, 32);

    float rinv[8];
    #pragma unroll
    for (int r = 0; r < 8; ++r)
        rinv[r] = 1.0f / sqrtf(fmaxf(fabsf(p[r]), 1e-8f));

    const size_t obase = (size_t)b * SEQ * DIM + (size_t)h * DPH;
    #pragma unroll
    for (int f = 0; f < 4; ++f)
        #pragma unroll
        for (int r = 0; r < 8; ++r)
            Out[obase + (size_t)(s0 + r + 8 * half) * DIM + f * 16 + l16] =
                acc[f][r] * rinv[r];
}

// ---------------------------------------------------------------------------
extern "C" void kernel_launch(void* const* d_in, const int* in_sizes, int n_in,
                              void* d_out, int out_size, void* d_ws, size_t ws_size,
                              hipStream_t stream)
{
    const float* key        = (const float*)d_in[0];
    const float* value      = (const float*)d_in[1];
    const float* query      = (const float*)d_in[2];
    const float* Wq         = (const float*)d_in[3];
    const float* bq         = (const float*)d_in[4];
    const float* lsq        = (const float*)d_in[5];
    const float* Wk         = (const float*)d_in[6];
    const float* bk         = (const float*)d_in[7];
    const float* lsk        = (const float*)d_in[8];
    const float* Wv         = (const float*)d_in[9];
    const float* bv         = (const float*)d_in[10];
    const float* lsv        = (const float*)d_in[11];
    const float* attn_scale = (const float*)d_in[12];
    const float* attn_bias  = (const float*)d_in[13];
    float* out = (float*)d_out;

    float* ws = (float*)d_ws;
    const size_t nproj = (size_t)NROWS * DIM;           // 8,388,608 floats each
    float* qbuf   = ws;
    float* kbuf   = ws + nproj;
    float* vbuf   = ws + 2 * nproj;
    float* Mpart  = ws + 3 * nproj;                     // 4*128*4096 floats
    float* vspart = Mpart + (size_t)4 * 128 * 4096;     // 4*128*64 floats
    __bf16* wpl   = (__bf16*)(vspart + (size_t)4 * 128 * 64);
    const size_t wsz = (size_t)DIM * DIM;               // 262144 elements
    __bf16* Wqh = wpl;            __bf16* Wql = wpl + wsz;
    __bf16* Wkh = wpl + 2 * wsz;  __bf16* Wkl = wpl + 3 * wsz;
    __bf16* Wvh = wpl + 4 * wsz;  __bf16* Wvl = wpl + 5 * wsz;

    dim3 blk(256);
    convert_w_kernel<<<wsz / 1024, blk, 0, stream>>>(Wq, Wqh, Wql);
    convert_w_kernel<<<wsz / 1024, blk, 0, stream>>>(Wk, Wkh, Wkl);
    convert_w_kernel<<<wsz / 1024, blk, 0, stream>>>(Wv, Wvh, Wvl);

    lorentz_proj_kernel<<<NROWS / 32, blk, 0, stream>>>(query, Wqh, Wql, bq, lsq, qbuf);
    lorentz_proj_kernel<<<NROWS / 32, blk, 0, stream>>>(key,   Wkh, Wkl, bk, lsk, kbuf);
    lorentz_proj_kernel<<<NROWS / 32, blk, 0, stream>>>(value, Wvh, Wvl, bv, lsv, vbuf);

    kv_moment_kernel<<<4 * BATCH * NH, blk, 0, stream>>>(kbuf, vbuf, Mpart, vspart);
    ctx_kernel<<<BATCH * NH * 8, blk, 0, stream>>>(qbuf, Mpart, vspart,
                                                   attn_scale, attn_bias, out);
}
